// BiDirCMLayer_25761213841554
// MI455X (gfx1250) — matrix operand order repaired and checked by reading, not compiled, not run on hardware
//
#include <hip/hip_runtime.h>
#include <math.h>

// ---------------------------------------------------------------------------
// BiDirCMLayer for MI455X (gfx1250, wave32, WMMA bf16 -> f32 accumulate,
// async global->LDS staging for pre-converted bf16 weight slabs)
// ---------------------------------------------------------------------------
#define LTXT 32
#define LVIS 8500
#define DIMD 256
#define NHEAD 8
#define HDIM 32
#define NLVL 4
#define NPTS 4
#define BATCH 8

typedef __attribute__((ext_vector_type(16))) __bf16 v16bf;
typedef __attribute__((ext_vector_type(8)))  float  v8f;

__device__ inline unsigned short f2bf_bits(float f) {
  union { float f; unsigned u; } v; v.f = f;
  unsigned r = v.u + 0x7FFFu + ((v.u >> 16) & 1u);   // round-to-nearest-even
  return (unsigned short)(r >> 16);
}
__device__ inline __bf16 f2bf(float f) {
  union { unsigned short s; __bf16 b; } o;
  o.s = f2bf_bits(f);
  return o.b;
}

// 16-byte async copy global -> LDS (GLOBAL_LOAD_ASYNC_TO_LDS_B128, ASYNCcnt)
__device__ inline void async_copy16(unsigned lds_off, const void* gsrc) {
  asm volatile("global_load_async_to_lds_b128 %0, %1, off"
               :: "v"(lds_off), "v"((unsigned long long)gsrc)
               : "memory");
}
__device__ inline void async_wait0() {
  asm volatile("s_wait_asynccnt 0" ::: "memory");
}

// A-fragment (16x32 bf16, row-major LDS tile): ISA 7.12.2 16-bit A layout.
__device__ inline v16bf load_fragA(const __bf16* __restrict__ base, int stride) {
  const int lane = threadIdx.x & 31;
  const __bf16* p = base + (lane & 15) * stride + ((lane >> 4) << 3);
  v16bf f;
#pragma unroll
  for (int j = 0; j < 4; ++j) {
    f[2*j]     = p[2*j];
    f[2*j+1]   = p[2*j+1];
    f[8+2*j]   = p[16+2*j];
    f[8+2*j+1] = p[16+2*j+1];
  }
  return f;
}

// B-fragment (32x16 bf16, N-major LDS tile [n][k]): ISA 7.12.2 16-bit layout (same k order as A).
__device__ inline v16bf load_fragB(const __bf16* __restrict__ base, int stride) {
  const int lane = threadIdx.x & 31;
  const __bf16* p = base + (lane & 15) * stride + ((lane >> 4) << 3);
  v16bf f;
#pragma unroll
  for (int j = 0; j < 16; ++j) f[j] = p[j < 8 ? j : j + 8];
  return f;
}

__device__ inline v8f wmma_bf16(v16bf a, v16bf b, v8f c) {
  return __builtin_amdgcn_wmma_f32_16x16x32_bf16(false, a, false, b, (short)0, c,
                                                 false, false);
}

// ---------------------------------------------------------------------------
// Weight pre-pass: out[n*K + k] = bf16(W[k*N + n])   (N-major / transposed)
// ---------------------------------------------------------------------------
__global__ void wtrans_bf16_kernel(const float* __restrict__ W, int K, int N,
                                   unsigned short* __restrict__ out) {
  long i = (long)blockIdx.x * 256 + threadIdx.x;
  if (i >= (long)K * N) return;
  int n = (int)(i / K), k = (int)(i % K);
  out[i] = f2bf_bits(W[(long)k * N + n]);
}

// ---------------------------------------------------------------------------
// Generic GEMM:  C[M,N] = A[M,K] @ W[K,N] + bias,  W row-major f32.
// A addressed as A[m*sAm + k*sAk] (row-major, or col-major NCHW feat gather).
// ---------------------------------------------------------------------------
__global__ __launch_bounds__(256) void gemm_bias_kernel(
    const float* __restrict__ A, long sAm, long sAk,
    const float* __restrict__ W, const float* __restrict__ bias,
    float* __restrict__ C, int M, int N, int K) {
  __shared__ __bf16 sA[16 * 34];
  __shared__ __bf16 sB[128 * 34];
  const int tid = threadIdx.x;
  const int wave = tid >> 5, lane = tid & 31;
  const int m0 = blockIdx.x * 16;
  const int n0 = blockIdx.y * 128;
  v8f acc = {};
  for (int k0 = 0; k0 < K; k0 += 32) {
    __syncthreads();
    if (sAk == 1) {
      for (int i = tid; i < 512; i += 256) {
        int mm = i >> 5, kk = i & 31;
        int mg = m0 + mm; if (mg >= M) mg = M - 1;
        sA[mm * 34 + kk] = f2bf(A[(long)mg * sAm + (long)(k0 + kk)]);
      }
    } else {
      for (int i = tid; i < 512; i += 256) {
        int mm = i & 15, kk = i >> 4;
        int mg = m0 + mm; if (mg >= M) mg = M - 1;
        sA[mm * 34 + kk] = f2bf(A[(long)mg * sAm + (long)(k0 + kk) * sAk]);
      }
    }
    for (int i = tid; i < 4096; i += 256) {
      int nn = i & 127, kk = i >> 7;
      sB[nn * 34 + kk] = f2bf(W[(long)(k0 + kk) * N + (n0 + nn)]);
    }
    if (k0 + 32 < K)
      __builtin_prefetch(W + (long)(k0 + 32) * N + n0 + (tid & 127), 0, 1);
    __syncthreads();
    v16bf af = load_fragA(sA, 34);
    v16bf bf = load_fragB(sB + (wave * 16) * 34, 34);
    acc = wmma_bf16(af, bf, acc);
  }
  const int col = n0 + wave * 16 + (lane & 15);
  const int rb  = (lane >> 4) * 8;
#pragma unroll
  for (int r = 0; r < 8; ++r) {
    int row = m0 + rb + r;
    if (row < M) C[(long)row * N + col] = acc[r] + bias[col];
  }
}

// ---------------------------------------------------------------------------
__global__ void add_kernel(const float* __restrict__ a, const float* __restrict__ b,
                           float* __restrict__ y, long n) {
  long i = (long)blockIdx.x * blockDim.x + threadIdx.x;
  if (i < n) y[i] = a[i] + b[i];
}

__global__ void softmax16_kernel(float* __restrict__ aw, int groups) {
  int i = blockIdx.x * blockDim.x + threadIdx.x;
  if (i >= groups) return;
  float* p = aw + (long)i * 16;
  float mx = p[0];
#pragma unroll
  for (int j = 1; j < 16; ++j) mx = fmaxf(mx, p[j]);
  float e[16], s = 0.f;
#pragma unroll
  for (int j = 0; j < 16; ++j) { e[j] = __expf(p[j] - mx); s += e[j]; }
  float inv = 1.f / s;
#pragma unroll
  for (int j = 0; j < 16; ++j) p[j] = e[j] * inv;
}

// out = LayerNorm(a + d) over last dim 256, 16 threads/row
__global__ __launch_bounds__(256) void resln_kernel(
    const float* __restrict__ a, const float* __restrict__ d,
    const float* __restrict__ g, const float* __restrict__ be,
    float* __restrict__ out, int rows) {
  const int tid = threadIdx.x;
  int row = blockIdx.x * 16 + (tid >> 4);
  const int sub = tid & 15;
  const bool wr = row < rows;
  if (row >= rows) row = rows - 1;
  const long base = (long)row * DIMD;
  float x[16], s = 0.f, sq = 0.f;
#pragma unroll
  for (int j = 0; j < 16; ++j) {
    int c = sub + j * 16;
    x[j] = a[base + c] + d[base + c];
    s += x[j]; sq += x[j] * x[j];
  }
#pragma unroll
  for (int m = 1; m < 16; m <<= 1) {
    s  += __shfl_xor(s, m, 32);
    sq += __shfl_xor(sq, m, 32);
  }
  float mean = s * (1.f / 256.f);
  float var  = sq * (1.f / 256.f) - mean * mean;
  float rstd = rsqrtf(var + 1e-5f);
  if (wr) {
#pragma unroll
    for (int j = 0; j < 16; ++j) {
      int c = sub + j * 16;
      out[base + c] = (x[j] - mean) * rstd * g[c] + be[c];
    }
  }
}

// ---------------------------------------------------------------------------
// Deformable multi-scale sampler: one wave per (b,q,h), lane = head dim.
// ---------------------------------------------------------------------------
__global__ __launch_bounds__(256) void deform_sample_kernel(
    const float* __restrict__ v, const float* __restrict__ off,
    const float* __restrict__ aw, const float* __restrict__ refp,
    float* __restrict__ samp) {
  const int gw = (blockIdx.x * 256 + threadIdx.x) >> 5;
  const int lane = threadIdx.x & 31;
  const int h = gw & 7;
  const int q = (gw >> 3) & 31;
  const int b = gw >> 8;
  const int Hs[4] = {80, 40, 20, 10};
  const int st[4] = {0, 6400, 8000, 8400};
  const long obase = (long)b * 32 + q;
  float acc = 0.f;
  for (int l = 0; l < 4; ++l) {
    const int Hl = Hs[l], Wl = Hs[l];
    const float rx = refp[(obase * 4 + l) * 2 + 0];
    const float ry = refp[(obase * 4 + l) * 2 + 1];
#pragma unroll
    for (int p = 0; p < 4; ++p) {
      const int oc = ((h * 4 + l) * 4 + p) * 2;
      const float ox = off[obase * 256 + oc];
      const float oy = off[obase * 256 + oc + 1];
      const float wA = aw[(obase * 8 + h) * 16 + l * 4 + p];
      const float x = (rx + ox / (float)Wl) * (float)Wl - 0.5f;
      const float y = (ry + oy / (float)Hl) * (float)Hl - 0.5f;
      const float x0f = floorf(x), y0f = floorf(y);
      const float fx = x - x0f, fy = y - y0f;
      const int x0 = (int)x0f, y0 = (int)y0f;
#pragma unroll
      for (int dy = 0; dy < 2; ++dy) {
#pragma unroll
        for (int dx = 0; dx < 2; ++dx) {
          const int xi = x0 + dx, yi = y0 + dy;
          const float wt = (dx ? fx : 1.f - fx) * (dy ? fy : 1.f - fy);
          if (xi >= 0 && xi < Wl && yi >= 0 && yi < Hl) {
            const long pos = st[l] + yi * Wl + xi;
            acc += wA * wt * v[((long)b * LVIS + pos) * 256 + h * 32 + lane];
          }
        }
      }
    }
  }
  samp[obase * 256 + h * 32 + lane] = acc;
}

// ---------------------------------------------------------------------------
// Fused FFN + residual + LayerNorm:  out = LN(x + gelu(x@W1+b1)@W2 + b2).
// 32 rows/block; hidden streamed in 128-wide chunks with GEMM2 accumulated on
// the fly (no 1024-wide buffer). Weight slabs: pre-converted bf16 N-major,
// staged via async global->LDS b128 copies. rows must be a multiple of 32.
// ---------------------------------------------------------------------------
__global__ __launch_bounds__(256) void ffn_ln_kernel(
    const float* __restrict__ X, float* __restrict__ Out, int rows,
    const unsigned short* __restrict__ W1T,  // bf16 [1024][256]
    const float* __restrict__ b1,
    const unsigned short* __restrict__ W2T,  // bf16 [256][1024]
    const float* __restrict__ b2,
    const float* __restrict__ g, const float* __restrict__ be) {
  __shared__ __align__(16) __bf16 sX[32 * 264];   // 16896 B
  __shared__ __align__(16) __bf16 sW[256 * 40];   // 20480 B (row stride 80 B)
  __shared__ __align__(16) __bf16 sH[32 * 136];   //  8704 B
  __shared__ float sStat[32][2];
  const int tid = threadIdx.x, wave = tid >> 5, lane = tid & 31;
  const int r0 = blockIdx.x * 32;
  for (int i = tid; i < 8192; i += 256) {
    int mm = i >> 8, cc = i & 255;
    int rg = r0 + mm; if (rg >= rows) rg = rows - 1;
    sX[mm * 264 + cc] = f2bf(X[(long)rg * 256 + cc]);
  }
  if (tid < 64) sStat[tid >> 1][tid & 1] = 0.f;

  const int mt  = wave >> 2;           // row tile (0/1) for both GEMMs
  const int co1 = (wave & 3) * 32;     // GEMM1 col base within 128-chunk
  const int cq2 = (wave & 3) * 64;     // GEMM2 col quarter of 256
  v8f y0 = {}, y1 = {}, y2 = {}, y3 = {};

  for (int nc = 0; nc < 1024; nc += 128) {
    // ---- GEMM1: h = x @ W1[:, nc:nc+128] ----
    v8f h0 = {}, h1 = {};
    for (int k0 = 0; k0 < 256; k0 += 32) {
      __syncthreads();
      for (int i = tid; i < 512; i += 256) {        // [128 n][32 k] slab
        int row = i >> 2, ch = i & 3;
        async_copy16((unsigned)(size_t)(sW + row * 40) + (unsigned)(ch * 16),
                     (const unsigned char*)(W1T + ((long)(nc + row) * 256 + k0)) + ch * 16);
      }
      async_wait0();
      __syncthreads();
      v16bf a  = load_fragA(sX + mt * 16 * 264 + k0, 264);
      v16bf w0 = load_fragB(sW + co1 * 40, 40);
      v16bf w1 = load_fragB(sW + (co1 + 16) * 40, 40);
      h0 = wmma_bf16(a, w0, h0);
      h1 = wmma_bf16(a, w1, h1);
    }
    // ---- bias + exact GELU -> sH ----
    {
      const int rb = (lane >> 4) * 8;
      const int c  = co1 + (lane & 15);
#pragma unroll
      for (int r = 0; r < 8; ++r) {
        int row = mt * 16 + rb + r;
        float v0 = h0[r] + b1[nc + c];
        float v1 = h1[r] + b1[nc + c + 16];
        v0 = 0.5f * v0 * (1.f + erff(v0 * 0.70710678118654752f));
        v1 = 0.5f * v1 * (1.f + erff(v1 * 0.70710678118654752f));
        sH[row * 136 + c]      = f2bf(v0);
        sH[row * 136 + c + 16] = f2bf(v1);
      }
    }
    // ---- GEMM2 partial: y += h_chunk @ W2[nc:nc+128, :] ----
    for (int k1 = 0; k1 < 128; k1 += 32) {
      __syncthreads();
      for (int i = tid; i < 1024; i += 256) {       // [256 n][32 k] slab
        int row = i >> 2, ch = i & 3;
        async_copy16((unsigned)(size_t)(sW + row * 40) + (unsigned)(ch * 16),
                     (const unsigned char*)(W2T + ((long)row * 1024 + nc + k1)) + ch * 16);
      }
      async_wait0();
      __syncthreads();
      v16bf a  = load_fragA(sH + mt * 16 * 136 + k1, 136);
      v16bf w0 = load_fragB(sW + (cq2 +  0) * 40, 40);
      v16bf w1 = load_fragB(sW + (cq2 + 16) * 40, 40);
      v16bf w2 = load_fragB(sW + (cq2 + 32) * 40, 40);
      v16bf w3 = load_fragB(sW + (cq2 + 48) * 40, 40);
      y0 = wmma_bf16(a, w0, y0);
      y1 = wmma_bf16(a, w1, y1);
      y2 = wmma_bf16(a, w2, y2);
      y3 = wmma_bf16(a, w3, y3);
    }
  }
  // ---- bias + residual + LN ----
  const int cB = cq2 + (lane & 15);
  const int rb = (lane >> 4) * 8;
  v8f yy[4] = {y0, y1, y2, y3};
  float yv[4][8];
#pragma unroll
  for (int r = 0; r < 8; ++r) {
    int row = mt * 16 + rb + r;
    int rg = r0 + row; int rc = rg < rows ? rg : rows - 1;
    const float* xr = X + (long)rc * 256;
    float s = 0.f, sq = 0.f;
#pragma unroll
    for (int j = 0; j < 4; ++j) {
      int c = cB + j * 16;
      float v = yy[j][r] + b2[c] + xr[c];
      yv[j][r] = v; s += v; sq += v * v;
    }
#pragma unroll
    for (int m = 1; m < 16; m <<= 1) {
      s  += __shfl_xor(s, m, 32);
      sq += __shfl_xor(sq, m, 32);
    }
    if ((lane & 15) == 0) {
      atomicAdd(&sStat[row][0], s);
      atomicAdd(&sStat[row][1], sq);
    }
  }
  __syncthreads();
#pragma unroll
  for (int r = 0; r < 8; ++r) {
    int row = mt * 16 + rb + r; int rg = r0 + row;
    float mean = sStat[row][0] * (1.f / 256.f);
    float var  = sStat[row][1] * (1.f / 256.f) - mean * mean;
    float rstd = rsqrtf(var + 1e-5f);
    if (rg < rows) {
#pragma unroll
      for (int j = 0; j < 4; ++j) {
        int c = cB + j * 16;
        Out[(long)rg * 256 + c] = (yv[j][r] - mean) * rstd * g[c] + be[c];
      }
    }
  }
}

// ---------------------------------------------------------------------------
// Fused vis->txt cross attention (16 vis rows/block, wave == head):
// q = (vis+pos)@Wq+bq; softmax(q.k^T/sqrt(32)) in WMMA accumulators;
// mo = p@v; y = mo@Wo+bo+vis; out_mid = LN1(y); out_attn = mean_h p.
// Wq/Wo consumed as pre-converted bf16 N-major via async LDS staging.
// ---------------------------------------------------------------------------
__global__ __launch_bounds__(256) void vli_attn_kernel(
    const float* __restrict__ vis_tokens, const float* __restrict__ vis_pos,
    const float* __restrict__ kh, const float* __restrict__ vh,   // [B][32][256]
    const unsigned short* __restrict__ WqT,  // bf16 [256][256]
    const float* __restrict__ bq,
    const unsigned short* __restrict__ WoT,  // bf16 [256][256]
    const float* __restrict__ bo,
    const float* __restrict__ g1, const float* __restrict__ be1,
    float* __restrict__ out_mid, float* __restrict__ out_attn) {
  __shared__ __align__(16) __bf16 sW[256 * 40];   // 20480 B
  __shared__ __bf16 sKV[256 * 34];                // sK[32][264] then sVT[256][34]
  __shared__ __bf16 sQ[16 * 264];                 // q tile, then mo tile
  __shared__ __bf16 sU1[4608];                    // xq [16][264], then p [8][16][36]
  __shared__ float  sPM[16 * 32];
  __shared__ float  sStat[16][2];
  const int tid = threadIdx.x, wave = tid >> 5, lane = tid & 31;
  const int b = blockIdx.y;
  const int r0 = blockIdx.x * 16;
  __bf16* sXq = sU1;
  __bf16* sK  = sKV;
  __bf16* sVT = sKV;
  __bf16* sMO = sQ;

  // phase 0: stage kh tile, (vis+pos) tile
  for (int i = tid; i < 8192; i += 256) {
    int t = i >> 8, c = i & 255;
    sK[t * 264 + c] = f2bf(kh[((long)b * 32 + t) * 256 + c]);
  }
  for (int i = tid; i < 4096; i += 256) {
    int mm = i >> 8, c = i & 255;
    int rg = r0 + mm; if (rg >= LVIS) rg = LVIS - 1;
    long a = ((long)b * LVIS + rg) * 256 + c;
    sXq[mm * 264 + c] = f2bf(vis_tokens[a] + vis_pos[a]);
  }
  for (int i = tid; i < 512; i += 256) sPM[i] = 0.f;
  if (tid < 32) sStat[tid >> 1][tid & 1] = 0.f;
  __syncthreads();

  // phase 1: q = xq @ Wq + bq -> sQ
  {
    v8f q0 = {}, q1 = {};
    for (int k0 = 0; k0 < 256; k0 += 32) {
      __syncthreads();
      for (int i = tid; i < 1024; i += 256) {
        int row = i >> 2, ch = i & 3;
        async_copy16((unsigned)(size_t)(sW + row * 40) + (unsigned)(ch * 16),
                     (const unsigned char*)(WqT + ((long)row * 256 + k0)) + ch * 16);
      }
      async_wait0();
      __syncthreads();
      v16bf a  = load_fragA(sXq + k0, 264);
      v16bf w0 = load_fragB(sW + (wave * 32) * 40, 40);
      v16bf w1 = load_fragB(sW + (wave * 32 + 16) * 40, 40);
      q0 = wmma_bf16(a, w0, q0);
      q1 = wmma_bf16(a, w1, q1);
    }
    const int c0 = wave * 32 + (lane & 15);
    const int rb = (lane >> 4) * 8;
#pragma unroll
    for (int r = 0; r < 8; ++r) {
      sQ[(rb + r) * 264 + c0]      = f2bf(q0[r] + bq[c0]);
      sQ[(rb + r) * 264 + c0 + 16] = f2bf(q1[r] + bq[c0 + 16]);
    }
  }
  __syncthreads();

  // phase 2: scores + register softmax (wave = head)
  const int h = wave;
  float p0[8], p1[8];
  {
    v16bf aq  = load_fragA(sQ + h * 32, 264);
    v16bf bk0 = load_fragB(sK + h * 32, 264);
    v16bf bk1 = load_fragB(sK + 16 * 264 + h * 32, 264);
    v8f zero = {};
    v8f s0 = wmma_bf16(aq, bk0, zero);
    v8f s1 = wmma_bf16(aq, bk1, zero);
    const float scale = 0.17677669529663687f;  // 1/sqrt(32)
#pragma unroll
    for (int r = 0; r < 8; ++r) {
      float a0 = s0[r] * scale, a1 = s1[r] * scale;
      float mx = fmaxf(a0, a1);
#pragma unroll
      for (int m = 1; m < 16; m <<= 1) mx = fmaxf(mx, __shfl_xor(mx, m, 32));
      float e0 = __expf(a0 - mx), e1 = __expf(a1 - mx);
      float sm = e0 + e1;
#pragma unroll
      for (int m = 1; m < 16; m <<= 1) sm += __shfl_xor(sm, m, 32);
      float inv = 1.f / sm;
      p0[r] = e0 * inv; p1[r] = e1 * inv;
    }
    const int tok = lane & 15;
    const int rb  = (lane >> 4) * 8;
    __bf16* sPw = sU1 + wave * 576;
#pragma unroll
    for (int r = 0; r < 8; ++r) {
      atomicAdd(&sPM[(rb + r) * 32 + tok],      p0[r] * 0.125f);
      atomicAdd(&sPM[(rb + r) * 32 + tok + 16], p1[r] * 0.125f);
      sPw[(rb + r) * 36 + tok]      = f2bf(p0[r]);
      sPw[(rb + r) * 36 + tok + 16] = f2bf(p1[r]);
    }
  }
  __syncthreads();

  // phase 3: emit head-averaged attn; stage vh transposed (sK is dead)
  const int valid = (LVIS - r0) < 16 ? (LVIS - r0) : 16;
  for (int i = tid; i < 512; i += 256) {
    int mm = i >> 5, tok = i & 31;
    if (mm < valid)
      out_attn[((long)b * LVIS + r0 + mm) * 32 + tok] = sPM[i];
  }
  for (int i = tid; i < 8192; i += 256) {
    int t = i >> 8, c = i & 255;
    sVT[c * 34 + t] = f2bf(vh[((long)b * 32 + t) * 256 + c]);
  }
  __syncthreads();

  // phase 4: mo = p @ v_head -> sMO
  {
    __bf16* sPw = sU1 + wave * 576;
    v16bf ap  = load_fragA(sPw, 36);
    v16bf bv0 = load_fragB(sVT + (h * 32) * 34, 34);
    v16bf bv1 = load_fragB(sVT + (h * 32 + 16) * 34, 34);
    v8f zero = {};
    v8f o0 = wmma_bf16(ap, bv0, zero);
    v8f o1 = wmma_bf16(ap, bv1, zero);
    const int c0 = h * 32 + (lane & 15);
    const int rb = (lane >> 4) * 8;
#pragma unroll
    for (int r = 0; r < 8; ++r) {
      sMO[(rb + r) * 264 + c0]      = f2bf(o0[r]);
      sMO[(rb + r) * 264 + c0 + 16] = f2bf(o1[r]);
    }
  }
  __syncthreads();

  // phase 5: y = mo @ Wo + bo + vis; LN1 -> out_mid
  v8f y0 = {}, y1 = {};
  for (int k0 = 0; k0 < 256; k0 += 32) {
    __syncthreads();
    for (int i = tid; i < 1024; i += 256) {
      int row = i >> 2, ch = i & 3;
      async_copy16((unsigned)(size_t)(sW + row * 40) + (unsigned)(ch * 16),
                   (const unsigned char*)(WoT + ((long)row * 256 + k0)) + ch * 16);
    }
    async_wait0();
    __syncthreads();
    v16bf a  = load_fragA(sMO + k0, 264);
    v16bf w0 = load_fragB(sW + (wave * 32) * 40, 40);
    v16bf w1 = load_fragB(sW + (wave * 32 + 16) * 40, 40);
    y0 = wmma_bf16(a, w0, y0);
    y1 = wmma_bf16(a, w1, y1);
  }
  {
    const int c0 = wave * 32 + (lane & 15);
    const int rb = (lane >> 4) * 8;
    float yv0[8], yv1[8];
#pragma unroll
    for (int r = 0; r < 8; ++r) {
      int rg = r0 + rb + r; int rc = rg < LVIS ? rg : LVIS - 1;
      long base = ((long)b * LVIS + rc) * 256;
      yv0[r] = y0[r] + bo[c0]      + vis_tokens[base + c0];
      yv1[r] = y1[r] + bo[c0 + 16] + vis_tokens[base + c0 + 16];
      float s  = yv0[r] + yv1[r];
      float sq = yv0[r] * yv0[r] + yv1[r] * yv1[r];
#pragma unroll
      for (int m = 1; m < 16; m <<= 1) {
        s  += __shfl_xor(s, m, 32);
        sq += __shfl_xor(sq, m, 32);
      }
      if ((lane & 15) == 0) {
        atomicAdd(&sStat[rb + r][0], s);
        atomicAdd(&sStat[rb + r][1], sq);
      }
    }
    __syncthreads();
#pragma unroll
    for (int r = 0; r < 8; ++r) {
      int row = rb + r; int rg = r0 + row;
      float mean = sStat[row][0] * (1.f / 256.f);
      float var  = sStat[row][1] * (1.f / 256.f) - mean * mean;
      float rstd = rsqrtf(var + 1e-5f);
      if (rg < LVIS) {
        long ob = ((long)b * LVIS + rg) * 256;
        out_mid[ob + c0]      = (yv0[r] - mean) * rstd * g1[c0] + be1[c0];
        out_mid[ob + c0 + 16] = (yv1[r] - mean) * rstd * g1[c0 + 16] + be1[c0 + 16];
      }
    }
  }
}

// ---------------------------------------------------------------------------
// host-side orchestration
// ---------------------------------------------------------------------------
extern "C" void kernel_launch(void* const* d_in, const int* in_sizes, int n_in,
                              void* d_out, int out_size, void* d_ws, size_t ws_size,
                              hipStream_t stream) {
  (void)in_sizes; (void)n_in; (void)out_size; (void)ws_size;
  const float* txt      = (const float*)d_in[0];
  const float* vis      = (const float*)d_in[1];
  const float* feat[4]  = {(const float*)d_in[2], (const float*)d_in[3],
                           (const float*)d_in[4], (const float*)d_in[5]};
  const float* refp     = (const float*)d_in[8];
  const float* txt_pos  = (const float*)d_in[9];
  const float* vis_pos  = (const float*)d_in[10];
  const float* W_off    = (const float*)d_in[11]; const float* b_off = (const float*)d_in[12];
  const float* W_aw     = (const float*)d_in[13]; const float* b_aw  = (const float*)d_in[14];
  const float* W_v      = (const float*)d_in[15]; const float* b_v   = (const float*)d_in[16];
  const float* W_do     = (const float*)d_in[17]; const float* b_do  = (const float*)d_in[18];
  const float* W_q_lvi  = (const float*)d_in[19]; const float* b_q_lvi = (const float*)d_in[20];
  const float* W_out_lvi= (const float*)d_in[21]; const float* b_out_lvi = (const float*)d_in[22];
  const float* ln1_g = (const float*)d_in[23]; const float* ln1_b = (const float*)d_in[24];
  const float* ln2_g = (const float*)d_in[25]; const float* ln2_b = (const float*)d_in[26];
  const float* ffn1_W1 = (const float*)d_in[27]; const float* ffn1_b1 = (const float*)d_in[28];
  const float* ffn1_W2 = (const float*)d_in[29]; const float* ffn1_b2 = (const float*)d_in[30];
  const float* Wq_m = (const float*)d_in[31]; const float* bq_m = (const float*)d_in[32];
  const float* Wk_m = (const float*)d_in[33]; const float* bk_m = (const float*)d_in[34];
  const float* Wv_m = (const float*)d_in[35]; const float* bv_m = (const float*)d_in[36];
  const float* Wo_m = (const float*)d_in[37]; const float* bo_m = (const float*)d_in[38];
  const float* vln1_g = (const float*)d_in[39]; const float* vln1_b = (const float*)d_in[40];
  const float* vln2_g = (const float*)d_in[41]; const float* vln2_b = (const float*)d_in[42];
  const float* ffn2_W1 = (const float*)d_in[43]; const float* ffn2_b1 = (const float*)d_in[44];
  const float* ffn2_W2 = (const float*)d_in[45]; const float* ffn2_b2 = (const float*)d_in[46];

  // workspace carve-up (float units)
  float* ws       = (float*)d_ws;
  float* w_v      = ws;                         // 17,408,000
  float* w_txtq   = w_v + 17408000L;
  float* w_q      = w_txtq + 65536;
  float* w_offo   = w_q + 65536;
  float* w_aw     = w_offo + 65536;
  float* w_samp   = w_aw + 32768;
  float* w_t1     = w_samp + 65536;
  float* w_t2     = w_t1 + 65536;
  float* w_txtkey = w_t2 + 65536;
  float* w_kh     = w_txtkey + 65536;
  float* w_vh     = w_kh + 65536;
  // bf16 pre-transposed weights (stored in float-sized slots)
  unsigned short* w_wqt   = (unsigned short*)(w_vh + 65536);      //  65,536 bf16
  unsigned short* w_wot   = (unsigned short*)(w_vh + 65536 + 32768);
  unsigned short* w_f1w1t = (unsigned short*)(w_vh + 65536 + 65536);   // 262,144 bf16
  unsigned short* w_f1w2t = (unsigned short*)(w_vh + 65536 + 196608);
  unsigned short* w_f2w1t = (unsigned short*)(w_vh + 65536 + 327680);
  unsigned short* w_f2w2t = (unsigned short*)(w_vh + 65536 + 458752);

  float* o_txt  = (float*)d_out;                // [8][32][256]
  float* o_vis  = o_txt + 65536;                // [8][8500][256]
  float* o_attn = o_vis + 17408000L;            // [8][8500][32]

  // ---- weight pre-pass: bf16 N-major ----
  wtrans_bf16_kernel<<<256,  256, 0, stream>>>(Wq_m,    256, 256,  w_wqt);
  wtrans_bf16_kernel<<<256,  256, 0, stream>>>(Wo_m,    256, 256,  w_wot);
  wtrans_bf16_kernel<<<1024, 256, 0, stream>>>(ffn1_W1, 256, 1024, w_f1w1t);
  wtrans_bf16_kernel<<<1024, 256, 0, stream>>>(ffn1_W2, 1024, 256, w_f1w2t);
  wtrans_bf16_kernel<<<1024, 256, 0, stream>>>(ffn2_W1, 256, 1024, w_f2w1t);
  wtrans_bf16_kernel<<<1024, 256, 0, stream>>>(ffn2_W2, 1024, 256, w_f2w2t);

  const dim3 g256(16, 2);

  // ---- lvi: text-side deformable attention ----
  add_kernel<<<256, 256, 0, stream>>>(txt, txt_pos, w_txtq, 65536);
  gemm_bias_kernel<<<g256, 256, 0, stream>>>(w_txtq, 256, 1, W_q_lvi, b_q_lvi, w_q, 256, 256, 256);
  gemm_bias_kernel<<<g256, 256, 0, stream>>>(w_q, 256, 1, W_off, b_off, w_offo, 256, 256, 256);
  gemm_bias_kernel<<<dim3(16, 1), 256, 0, stream>>>(w_q, 256, 1, W_aw, b_aw, w_aw, 256, 128, 256);
  softmax16_kernel<<<8, 256, 0, stream>>>(w_aw, 2048);

  const int Hs[4] = {80, 40, 20, 10};
  const int st[4] = {0, 6400, 8000, 8400};
  for (int b = 0; b < BATCH; ++b) {
    for (int l = 0; l < NLVL; ++l) {
      const int hw = Hs[l] * Hs[l];
      const float* A = feat[l] + (long)b * DIMD * hw;
      float* C = w_v + ((long)b * LVIS + st[l]) * DIMD;
      gemm_bias_kernel<<<dim3((hw + 15) / 16, 2), 256, 0, stream>>>(
          A, 1, hw, W_v, b_v, C, hw, 256, 256);
    }
  }
  deform_sample_kernel<<<256, 256, 0, stream>>>(w_v, w_offo, w_aw, refp, w_samp);
  gemm_bias_kernel<<<g256, 256, 0, stream>>>(w_samp, 256, 1, W_do, b_do, w_t1, 256, 256, 256);
  gemm_bias_kernel<<<g256, 256, 0, stream>>>(w_t1, 256, 1, W_out_lvi, b_out_lvi, w_t2, 256, 256, 256);
  resln_kernel<<<16, 256, 0, stream>>>(txt, w_t2, ln1_g, ln1_b, o_txt, 256);
  ffn_ln_kernel<<<8, 256, 0, stream>>>(o_txt, o_txt, 256,
                                       w_f1w1t, ffn1_b1, w_f1w2t, ffn1_b2, ln2_g, ln2_b);

  // ---- vli: vis-side cross attention over 32 text keys ----
  add_kernel<<<256, 256, 0, stream>>>(o_txt, txt_pos, w_txtkey, 65536);
  gemm_bias_kernel<<<g256, 256, 0, stream>>>(w_txtkey, 256, 1, Wk_m, bk_m, w_kh, 256, 256, 256);
  gemm_bias_kernel<<<g256, 256, 0, stream>>>(w_txtkey, 256, 1, Wv_m, bv_m, w_vh, 256, 256, 256);
  vli_attn_kernel<<<dim3(532, 8), 256, 0, stream>>>(vis, vis_pos, w_kh, w_vh,
                                                    w_wqt, bq_m, w_wot, bo_m,
                                                    vln1_g, vln1_b, o_vis, o_attn);
  ffn_ln_kernel<<<2125, 256, 0, stream>>>(o_vis, o_vis, 68000,
                                          w_f2w1t, ffn2_b1, w_f2w2t, ffn2_b2, vln2_g, vln2_b);
}
